// UBlock_51548197487206
// MI455X (gfx1250) — matrix operand order repaired and checked by reading, not compiled, not run on hardware
//
#include <hip/hip_runtime.h>

// ============================================================================
// Sparse-conv UBlock for MI455X (gfx1250, wave32, WMMA + TDM).
// Strategy: bf16 activations/weights + v_wmma_f32_16x16x32_bf16, f32 accum,
// LDS-staged gather, TDM (tensor_load_to_lds) weight staging,
// global_atomic_add_f32 scatter. Rulebooks synthesized on-device
// (reference rulebooks are host-side RNG constants, not inputs).
// ============================================================================

typedef unsigned short u16;
typedef __attribute__((ext_vector_type(16))) __bf16 v16bf;
typedef __attribute__((ext_vector_type(8)))  float  v8f;
typedef __attribute__((ext_vector_type(4)))  unsigned int v4u;
typedef __attribute__((ext_vector_type(8)))  int    v8i;
typedef __attribute__((ext_vector_type(4)))  int    v4i;

#define BN_EPS 1e-5f

__device__ __forceinline__ u16 f2bf(float f) {
  unsigned u = __float_as_uint(f);
  return (u16)((u + (((u >> 16) & 1u) + 0x7fffu)) >> 16);  // RNE
}

// ---------------------------------------------------------------------------
// setup kernels
// ---------------------------------------------------------------------------
__global__ void k_iota(int* __restrict__ p, int n) {
  int i = blockIdx.x * blockDim.x + threadIdx.x;
  if (i < n) p[i] = i;
}

// Synthetic rulebook: rin = hash(p,k) % n_in (gather), rout = p % n_out.
__global__ void k_init_rules(int* __restrict__ rin, int* __restrict__ rout,
                             int npairs, int n_off, int n_in, int n_out) {
  long long i = (long long)blockIdx.x * blockDim.x + threadIdx.x;
  long long tot = (long long)npairs * n_off;
  if (i >= tot) return;
  int k = (int)(i / npairs);
  int p = (int)(i % npairs);
  unsigned h = ((unsigned)p * 2654435761u) ^ ((unsigned)(k + 1) * 0x9E3779B9u);
  rin[i]  = (int)(h % (unsigned)n_in);
  rout[i] = p % n_out;
}

// f32 [n_off][cin][cout] -> bf16 [n_off][cout][cin] (B-fragment friendly)
__global__ void k_convert_w(const float* __restrict__ src, u16* __restrict__ dst,
                            int n_off, int cin, int cout) {
  int i = blockIdx.x * blockDim.x + threadIdx.x;
  int per = cin * cout;
  if (i >= n_off * per) return;
  int k = i / per, r = i % per, ci = r / cout, co = r % cout;
  dst[((size_t)k * cout + co) * cin + ci] = f2bf(src[i]);
}

__global__ void k_to_bf16(const float* __restrict__ x, u16* __restrict__ y, int n) {
  int i = blockIdx.x * blockDim.x + threadIdx.x;
  if (i < n) y[i] = f2bf(x[i]);
}

__global__ void k_add_inplace(float* __restrict__ d, const float* __restrict__ s, int n) {
  int i = blockIdx.x * blockDim.x + threadIdx.x;
  if (i < n) d[i] += s[i];
}

__global__ void k_add_out(float* __restrict__ o, const float* __restrict__ a,
                          const float* __restrict__ b, int n) {
  int i = blockIdx.x * blockDim.x + threadIdx.x;
  if (i < n) o[i] = a[i] + b[i];
}

__global__ void k_concat(float* __restrict__ cat, const float* __restrict__ a,
                         const float* __restrict__ b, int n_rows) {
  int i = blockIdx.x * blockDim.x + threadIdx.x;
  if (i >= n_rows * 64) return;
  int r = i >> 6, c = i & 63;
  cat[i] = (c < 32) ? a[r * 32 + c] : b[r * 32 + (c - 32)];
}

// ---------------------------------------------------------------------------
// BatchNorm (training-style: batch mean/var) + ReLU -> bf16 activations
// ---------------------------------------------------------------------------
template<int C>
__global__ __launch_bounds__(256)
void bn_stats(const float* __restrict__ x, float* __restrict__ stats, int n) {
  __shared__ float ls[512];
  const int t = threadIdx.x;
  const int c = t % C;
  const int slot = t / C;
  constexpr int SLOTS = 256 / C;
  float s = 0.f, q = 0.f;
  for (int r = blockIdx.x * SLOTS + slot; r < n; r += gridDim.x * SLOTS) {
    float v = x[(size_t)r * C + c];
    s += v; q += v * v;
  }
  ls[t] = s; ls[256 + t] = q;
  __syncthreads();
  if (t < C) {
    float ts = 0.f, tq = 0.f;
#pragma unroll
    for (int j = 0; j < SLOTS; ++j) { ts += ls[j * C + t]; tq += ls[256 + j * C + t]; }
    unsafeAtomicAdd(&stats[t], ts);        // global_atomic_add_f32
    unsafeAtomicAdd(&stats[C + t], tq);
  }
}

template<int C>
__global__ void bn_relu_bf16(const float* __restrict__ x, const float* __restrict__ stats,
                             const float* __restrict__ g, const float* __restrict__ b,
                             u16* __restrict__ y, int n) {
  int i = blockIdx.x * blockDim.x + threadIdx.x;
  if (i >= n * C) return;
  int c = i % C;
  float inv_n = 1.f / (float)n;
  float m   = stats[c] * inv_n;
  float var = stats[C + c] * inv_n - m * m;
  float sc  = rsqrtf(var + BN_EPS) * g[c];
  float sh  = b[c] - m * sc;
  y[i] = f2bf(fmaxf(x[i] * sc + sh, 0.f));
}

// ---------------------------------------------------------------------------
// Fused gather -> WMMA GEMM (bf16, f32 acc) -> scatter-add
// Block: 256 threads (8 wave32), TM=128 pairs per block, gridDim.y = offset.
// Weight tile is DMA'd into LDS by the Tensor Data Mover (wave 0 issues
// tensor_load_to_lds, waits on TENSORcnt, then the workgroup barrier
// publishes the tile) while all waves gather activation rows.
// A frag (16x32 bf16): lanes 0-15 hold M=0..15; VGPR0-3 = K(half*8..+7),
//   VGPR4-7 = K(16+half*8..+7).  B frag (32x16): lane n=l%16, K run
//   half*8..+7 and 16+half*8..+7 (as A) in Wt[cout][cin].  C: M = v + 8*half, N = l%16.
// ---------------------------------------------------------------------------
template<int CIN, int COUT>
__global__ __launch_bounds__(256)
void sparse_conv_wmma(const u16* __restrict__ act,   // [n_in][CIN] bf16
                      const u16* __restrict__ Wt,    // [n_off][COUT][CIN] bf16
                      const int* __restrict__ rin,   // [n_off][npairs]
                      const int* __restrict__ rout,  // [n_off][npairs]
                      float* __restrict__ out,       // [n_out][COUT] f32 (pre-zeroed)
                      int npairs) {
  constexpr int TM = 128;
  __shared__ __align__(16) u16 Alds[TM * CIN];
  __shared__ __align__(16) u16 Wlds[COUT * CIN];

  const int off = blockIdx.y;
  const long long offBase = (long long)off * npairs;
  const int tileBase = blockIdx.x * TM;

  // ---- stage weights for this offset via the Tensor Data Mover ----
#if __has_builtin(__builtin_amdgcn_tensor_load_to_lds) && __has_builtin(__builtin_amdgcn_s_wait_tensorcnt)
  if (threadIdx.x < 32) {   // wave 0 issues the DMA; TENSORcnt is per-wave
    const unsigned ldsAddr = (unsigned)(size_t)(void*)Wlds;
    const unsigned long long ga =
        (unsigned long long)(size_t)(Wt + (size_t)off * COUT * CIN);
    const unsigned ndw = (unsigned)(COUT * CIN / 2);  // tile = 1 row of dwords
    // D# group 0: count=1 | lds_addr | global_addr[56:0] | type=2
    v4u g0;
    g0.x = 1u;
    g0.y = ldsAddr;
    g0.z = (unsigned)(ga & 0xffffffffu);
    g0.w = (unsigned)((ga >> 32) & 0x01ffffffu) | 0x80000000u;
    // D# group 1: data_size=4B; tensor_dim0=ndw; tensor_dim1=1;
    //             tile_dim0=ndw; tile_dim1=1; dim0_stride=ndw
    v8i g1;
    g1[0] = (int)(2u << 16);                 // workgroup_mask=0, data_size=2 (4B)
    g1[1] = (int)((ndw & 0xffffu) << 16);    // tensor_dim0[15:0] @ bits 63:48
    g1[2] = (int)((ndw >> 16) | (1u << 16)); // tensor_dim0[31:16] | tensor_dim1=1
    g1[3] = (int)(ndw << 16);                // tile_dim0 @ bits 127:112
    g1[4] = 1;                               // tile_dim1 = 1
    g1[5] = (int)ndw;                        // tensor_dim0_stride[31:0]
    g1[6] = 0;
    g1[7] = 0;
    v4i gz4 = {0, 0, 0, 0};                  // groups 2/3 unused (<=2D tensor)
    v8i gz8 = {0, 0, 0, 0, 0, 0, 0, 0};
    __builtin_amdgcn_tensor_load_to_lds(g0, g1, gz4, gz4, gz8, 0);
    __builtin_amdgcn_s_wait_tensorcnt(0);
  }
#else
  {
    const uint4* src = (const uint4*)(Wt + (size_t)off * COUT * CIN);
    uint4* dst = (uint4*)Wlds;
    for (int i = threadIdx.x; i < COUT * CIN / 8; i += 256) dst[i] = src[i];
  }
#endif

  // ---- gather 128 activation rows (bf16) into LDS (overlaps the DMA) ----
  constexpr int CHR = CIN / 8;  // uint4 chunks per row
  for (int ch = threadIdx.x; ch < TM * CHR; ch += 256) {
    int r = ch / CHR, s = ch % CHR;
    int p = tileBase + r;
    uint4 val = {0u, 0u, 0u, 0u};
    if (p < npairs) {
      int srcRow = rin[offBase + p];
      val = *(const uint4*)(act + (size_t)srcRow * CIN + s * 8);
    }
    *(uint4*)(&Alds[r * CIN + s * 8]) = val;
  }
  if (threadIdx.x == 0 && tileBase + TM < npairs)
    __builtin_prefetch(rin + offBase + tileBase + TM, 0, 1);  // global_prefetch_b8
  __syncthreads();

  const int wave = threadIdx.x >> 5;
  const int lane = threadIdx.x & 31;
  const int hv = lane >> 4;      // lane half selects K sub-range
  const int lm = lane & 15;      // M (for A) / N (for B,C)

  union Frag { v16bf v; u16 u[16]; uint4 q[2]; };

  // A fragments for this wave's 16 rows (reused across all N tiles)
  Frag afr[CIN / 32];
#pragma unroll
  for (int kt = 0; kt < CIN / 32; ++kt) {
    const u16* arow = &Alds[(wave * 16 + lm) * CIN + kt * 32];
    afr[kt].q[0] = *(const uint4*)(arow + hv * 8);
    afr[kt].q[1] = *(const uint4*)(arow + 16 + hv * 8);
  }

#pragma unroll
  for (int nt = 0; nt < COUT / 16; ++nt) {
    v8f c = {0.f, 0.f, 0.f, 0.f, 0.f, 0.f, 0.f, 0.f};
#pragma unroll
    for (int kt = 0; kt < CIN / 32; ++kt) {
      Frag b;
      const u16* brow = &Wlds[(nt * 16 + lm) * CIN + kt * 32 + hv * 8];
      b.q[0] = *(const uint4*)(brow);
      b.q[1] = *(const uint4*)(brow + 16);
      c = __builtin_amdgcn_wmma_f32_16x16x32_bf16(
          false, afr[kt].v, false, b.v, (short)0, c, false, false);
    }
    // scatter-add: M = v + 8*hv within wave tile, N = lm
#pragma unroll
    for (int v = 0; v < 8; ++v) {
      int p = tileBase + wave * 16 + v + 8 * hv;
      if (p < npairs) {
        int dstRow = rout[offBase + p];
        unsafeAtomicAdd(&out[(size_t)dstRow * COUT + nt * 16 + lm], c[v]);
      }
    }
  }
}

// ---------------------------------------------------------------------------
// host orchestration
// ---------------------------------------------------------------------------
static inline int cdiv(long long a, long long b) { return (int)((a + b - 1) / b); }

extern "C" void kernel_launch(void* const* d_in, const int* in_sizes, int n_in,
                              void* d_out, int out_size, void* d_ws, size_t ws_size,
                              hipStream_t stream) {
  (void)n_in; (void)out_size; (void)ws_size;
  const int N1 = in_sizes[0] / 32;                       // 300000
  const int N2 = (int)(((long long)N1 * 62) / 100);      // ~ stride-2 active sites

  const float* feats  = (const float*)d_in[0];
  const float* W33_l1 = (const float*)d_in[1];
  const float* W_t0w1 = (const float*)d_in[2];
  const float* Wi_t0  = (const float*)d_in[3];
  const float* W33_l2 = (const float*)d_in[4];
  const float* W_down = (const float*)d_in[5];
  const float* W_up   = (const float*)d_in[6];
  const float* g32 = (const float*)d_in[7];
  const float* b32 = (const float*)d_in[8];
  const float* g64 = (const float*)d_in[9];
  const float* b64 = (const float*)d_in[10];

  // -------- workspace carve --------
  char* base = (char*)d_ws;
  size_t woff = 0;
  auto carve = [&](size_t bytes) -> char* {
    char* p = base + woff;
    woff += (bytes + 255) & ~(size_t)255;
    return p;
  };
  u16*   act   = (u16*)  carve((size_t)N1 * 64 * 2);
  float* h1    = (float*)carve((size_t)N1 * 32 * 4);
  float* h2    = (float*)carve((size_t)N1 * 32 * 4);
  float* hB    = (float*)carve((size_t)N1 * 32 * 4);
  float* cat   = (float*)carve((size_t)N1 * 64 * 4);
  float* D1    = (float*)carve((size_t)N2 * 64 * 4);
  float* D2    = (float*)carve((size_t)N2 * 64 * 4);
  float* D3    = (float*)carve((size_t)N2 * 64 * 4);
  float* stats = (float*)carve(2 * 64 * 4);

  const size_t n_l1 = (size_t)7 * 27 * 32 * 32, n_t0 = (size_t)27 * 64 * 32;
  const size_t n_wi = 64 * 32, n_l2 = (size_t)4 * 27 * 64 * 64;
  const size_t n_dn = (size_t)8 * 32 * 64, n_up = (size_t)8 * 64 * 32;
  u16* w_l1 = (u16*)carve(n_l1 * 2);
  u16* w_t0 = (u16*)carve(n_t0 * 2);
  u16* w_wi = (u16*)carve(n_wi * 2);
  u16* w_l2 = (u16*)carve(n_l2 * 2);
  u16* w_dn = (u16*)carve(n_dn * 2);
  u16* w_up = (u16*)carve(n_up * 2);

  int* s1_in  = (int*)carve((size_t)27 * N1 * 4);
  int* s1_out = (int*)carve((size_t)27 * N1 * 4);
  int* s2_in  = (int*)carve((size_t)27 * N2 * 4);
  int* s2_out = (int*)carve((size_t)27 * N2 * 4);
  int* dn_in  = (int*)carve((size_t)8 * N2 * 4);
  int* dn_out = (int*)carve((size_t)8 * N2 * 4);
  int* idN1   = (int*)carve((size_t)N1 * 4);

  // -------- per-call setup (deterministic, no caching) --------
  k_convert_w<<<cdiv(n_l1, 256), 256, 0, stream>>>(W33_l1, w_l1, 7 * 27, 32, 32);
  k_convert_w<<<cdiv(n_t0, 256), 256, 0, stream>>>(W_t0w1, w_t0, 27, 64, 32);
  k_convert_w<<<cdiv(n_wi, 256), 256, 0, stream>>>(Wi_t0,  w_wi, 1, 64, 32);
  k_convert_w<<<cdiv(n_l2, 256), 256, 0, stream>>>(W33_l2, w_l2, 4 * 27, 64, 64);
  k_convert_w<<<cdiv(n_dn, 256), 256, 0, stream>>>(W_down, w_dn, 8, 32, 64);
  k_convert_w<<<cdiv(n_up, 256), 256, 0, stream>>>(W_up,   w_up, 8, 64, 32);
  k_init_rules<<<cdiv((long long)27 * N1, 256), 256, 0, stream>>>(s1_in, s1_out, N1, 27, N1, N1);
  k_init_rules<<<cdiv((long long)27 * N2, 256), 256, 0, stream>>>(s2_in, s2_out, N2, 27, N2, N2);
  k_init_rules<<<cdiv((long long)8  * N2, 256), 256, 0, stream>>>(dn_in, dn_out, N2, 8, N1, N2);
  k_iota<<<cdiv(N1, 256), 256, 0, stream>>>(idN1, N1);

  auto bnrelu = [&](const float* x, int n, int C, const float* g, const float* b) {
    (void)hipMemsetAsync(stats, 0, 2 * C * sizeof(float), stream);
    if (C == 32) {
      bn_stats<32><<<512, 256, 0, stream>>>(x, stats, n);
      bn_relu_bf16<32><<<cdiv((long long)n * 32, 256), 256, 0, stream>>>(x, stats, g, b, act, n);
    } else {
      bn_stats<64><<<512, 256, 0, stream>>>(x, stats, n);
      bn_relu_bf16<64><<<cdiv((long long)n * 64, 256), 256, 0, stream>>>(x, stats, g, b, act, n);
    }
  };
  auto conv = [&](int cin, int cout, const u16* wt, const int* rin, const int* rout,
                  int n_off, int npairs, float* outp, int n_out) {
    (void)hipMemsetAsync(outp, 0, (size_t)n_out * cout * sizeof(float), stream);
    dim3 grid(cdiv(npairs, 128), n_off);
    if (cin == 32 && cout == 32)
      sparse_conv_wmma<32, 32><<<grid, 256, 0, stream>>>(act, wt, rin, rout, outp, npairs);
    else if (cin == 32 && cout == 64)
      sparse_conv_wmma<32, 64><<<grid, 256, 0, stream>>>(act, wt, rin, rout, outp, npairs);
    else if (cin == 64 && cout == 64)
      sparse_conv_wmma<64, 64><<<grid, 256, 0, stream>>>(act, wt, rin, rout, outp, npairs);
    else
      sparse_conv_wmma<64, 32><<<grid, 256, 0, stream>>>(act, wt, rin, rout, outp, npairs);
  };
  auto addip = [&](float* d, const float* s, int n) {
    k_add_inplace<<<cdiv(n, 256), 256, 0, stream>>>(d, s, n);
  };

  const size_t sl1 = (size_t)27 * 32 * 32;   // per-layer stride in w_l1
  const size_t sl2 = (size_t)27 * 64 * 64;   // per-layer stride in w_l2

  // ---- L1 residual block 1 ----
  bnrelu(feats, N1, 32, g32 + 0 * 32, b32 + 0 * 32);
  conv(32, 32, w_l1 + 0 * sl1, s1_in, s1_out, 27, N1, h1, N1);
  bnrelu(h1, N1, 32, g32 + 1 * 32, b32 + 1 * 32);
  conv(32, 32, w_l1 + 1 * sl1, s1_in, s1_out, 27, N1, h2, N1);
  addip(h2, feats, N1 * 32);                               // x1 = h2
  // ---- L1 residual block 2 ----
  bnrelu(h2, N1, 32, g32 + 2 * 32, b32 + 2 * 32);
  conv(32, 32, w_l1 + 2 * sl1, s1_in, s1_out, 27, N1, h1, N1);
  bnrelu(h1, N1, 32, g32 + 3 * 32, b32 + 3 * 32);
  conv(32, 32, w_l1 + 3 * sl1, s1_in, s1_out, 27, N1, hB, N1);
  addip(hB, h2, N1 * 32);                                  // identity = hB
  // ---- downsample 32->64 ----
  bnrelu(hB, N1, 32, g32 + 4 * 32, b32 + 4 * 32);
  conv(32, 64, w_dn, dn_in, dn_out, 8, N2, D1, N2);
  // ---- L2 residual block 1 ----
  bnrelu(D1, N2, 64, g64 + 0 * 64, b64 + 0 * 64);
  conv(64, 64, w_l2 + 0 * sl2, s2_in, s2_out, 27, N2, D2, N2);
  bnrelu(D2, N2, 64, g64 + 1 * 64, b64 + 1 * 64);
  conv(64, 64, w_l2 + 1 * sl2, s2_in, s2_out, 27, N2, D3, N2);
  addip(D3, D1, N2 * 64);
  // ---- L2 residual block 2 ----
  bnrelu(D3, N2, 64, g64 + 2 * 64, b64 + 2 * 64);
  conv(64, 64, w_l2 + 2 * sl2, s2_in, s2_out, 27, N2, D2, N2);
  bnrelu(D2, N2, 64, g64 + 3 * 64, b64 + 3 * 64);
  conv(64, 64, w_l2 + 3 * sl2, s2_in, s2_out, 27, N2, D1, N2);
  addip(D1, D3, N2 * 64);
  // ---- upsample 64->32 (transposed DOWN rulebook: swap in/out) ----
  bnrelu(D1, N2, 64, g64 + 4 * 64, b64 + 4 * 64);
  conv(64, 32, w_up, dn_out, dn_in, 8, N2, h1, N1);        // u = h1
  // ---- skip concat [identity | u] ----
  k_concat<<<cdiv((long long)N1 * 64, 256), 256, 0, stream>>>(cat, hB, h1, N1);
  // ---- tail block 1 (64->32, Wi projection on raw input) ----
  k_to_bf16<<<cdiv((long long)N1 * 64, 256), 256, 0, stream>>>(cat, act, N1 * 64);
  conv(64, 32, w_wi, idN1, idN1, 1, N1, h2, N1);           // identW = h2
  bnrelu(cat, N1, 64, g64 + 5 * 64, b64 + 5 * 64);
  conv(64, 32, w_t0, s1_in, s1_out, 27, N1, h1, N1);
  bnrelu(h1, N1, 32, g32 + 5 * 32, b32 + 5 * 32);
  conv(32, 32, w_l1 + 4 * sl1, s1_in, s1_out, 27, N1, hB, N1);
  addip(hB, h2, N1 * 32);                                  // x3 = hB
  // ---- tail block 2 (32->32) ----
  bnrelu(hB, N1, 32, g32 + 6 * 32, b32 + 6 * 32);
  conv(32, 32, w_l1 + 5 * sl1, s1_in, s1_out, 27, N1, h1, N1);
  bnrelu(h1, N1, 32, g32 + 7 * 32, b32 + 7 * 32);
  conv(32, 32, w_l1 + 6 * sl1, s1_in, s1_out, 27, N1, h2, N1);
  k_add_out<<<cdiv((long long)N1 * 32, 256), 256, 0, stream>>>((float*)d_out, h2, hB, N1 * 32);
}
